// Relation_69965017252118
// MI455X (gfx1250) — compile-verified, matrix operand order checked by reading, not run on hardware
//
#include <hip/hip_runtime.h>
#include <hip/hip_bf16.h>

// ---------------------------------------------------------------------------
// Neighborhood attention (5x5 window, 8 heads) on MI455X / gfx1250.
//   Stage 0: one-shot convert w_qkv (768x128 fp32) -> bf16 in workspace.
//   Stage 1: QKV projection GEMM  W(768x128) x X(128x16384) via
//            v_wmma_f32_16x16x32_bf16 (bf16 in, fp32 accumulate).
//            Block = one 16-pixel tile, all 768 out-channels; the x-tile is
//            staged once to LDS as bf16 and shared by all 8 waves (B frags
//            from ds_load_b128); A frags load pre-converted bf16 directly.
//   Stage 2: per-(pixel,head) neighborhood attention, fp32 VALU + v_exp_f32.
// ---------------------------------------------------------------------------

typedef __attribute__((ext_vector_type(16))) __bf16 v16bf;
typedef __attribute__((ext_vector_type(8)))  __bf16 v8bf;
typedef __attribute__((ext_vector_type(8)))  float  v8f;

#define NB    4
#define CIN   128
#define COUT  768
#define DMOD  256
#define HH    64
#define WW    64
#define HWPX  (HH * WW)          // 4096
#define NPIX  (NB * HWPX)        // 16384
#define NHEADS 8
#define HDIM  32
#define LDSROW 136               // bf16 per pixel row: 272B = 17*16 -> 16B
                                 // aligned rows + 4-bank skew per pixel

// ---------------------------------------------------------------------------
// Stage 0: w_qkv fp32 -> bf16, row-major [o][c]. 98304 elems, 4 per thread.
// ---------------------------------------------------------------------------
__global__ __launch_bounds__(256) void wcvt_kernel(
    const float* __restrict__ w, __bf16* __restrict__ wbf) {
  const int i = (blockIdx.x * 256 + threadIdx.x) * 4;   // < 98304
  const float4 f = *(const float4*)(w + i);
  v8bf* dst4 = (v8bf*)nullptr; (void)dst4;
  __bf16 v[4] = {(__bf16)f.x, (__bf16)f.y, (__bf16)f.z, (__bf16)f.w};
  *(uint64_t*)(wbf + i) = *(const uint64_t*)v;          // 8B store (4 x bf16)
}

// ---------------------------------------------------------------------------
// Stage 1: QKV GEMM.
//   grid = 1024 blocks (one per 16-pixel tile), 256 threads = 8 waves.
//   Wave w computes M-tiles m0 = (g*8 + w)*16 for g = 0..5  (48 tiles total).
// ---------------------------------------------------------------------------
__global__ __launch_bounds__(256) void qkv_gemm_wmma(
    const float* __restrict__ x, const __bf16* __restrict__ wbf,
    float* __restrict__ qkv) {
  __shared__ __bf16 sB[16 * LDSROW];

  const int tidx = threadIdx.x;
  const int p0   = blockIdx.x << 4;              // first pixel of tile

  // ---- stage x-tile (16 px * 128 ch) -> LDS bf16, [pixel][K] transposed ----
  {
    const int c   = tidx >> 1;                   // channel 0..127
    const int pg  = (tidx & 1) << 3;             // pixel sub-group 0 / 8
    const int n   = p0 >> 12;                    // tile never straddles n
    const int rem = (p0 & 4095) + pg;
    const float4* __restrict__ src =
        (const float4*)(x + (size_t)n * (CIN * HWPX) + (size_t)c * HWPX + rem);
    const float4 f0 = src[0];
    const float4 f1 = src[1];
    const __bf16 v[8] = {(__bf16)f0.x, (__bf16)f0.y, (__bf16)f0.z, (__bf16)f0.w,
                         (__bf16)f1.x, (__bf16)f1.y, (__bf16)f1.z, (__bf16)f1.w};
#pragma unroll
    for (int e = 0; e < 8; ++e) sB[(pg + e) * LDSROW + c] = v[e];
  }
  __syncthreads();

  const int wave  = tidx >> 5;
  const int lane  = tidx & 31;
  const int lmod  = lane & 15;
  const int lhalf = lane >> 4;
  const int koff  = lhalf << 3;                  // ISA A/B fragment K sub-base

  // ---- preload the 4 B fragments (K = 0..127) once; reuse for 6 M-tiles ----
  v16bf bfrag[4];
#pragma unroll
  for (int kb = 0; kb < 4; ++kb) {
    const __bf16* __restrict__ bp = &sB[lmod * LDSROW + kb * 32 + koff];
    const v8bf lo = *(const v8bf*)bp;            // K = kb*32 + koff + 0..7
    const v8bf hi = *(const v8bf*)(bp + 16);     // K = kb*32 + koff + 16..23
    bfrag[kb] = __builtin_shufflevector(lo, hi, 0, 1, 2, 3, 4, 5, 6, 7,
                                        8, 9, 10, 11, 12, 13, 14, 15);
  }

  const int pcol = p0 + lmod;                    // this lane's output pixel
#pragma unroll
  for (int g = 0; g < 6; ++g) {
    const int m0   = ((g << 3) + wave) << 4;     // M-tile base channel
    const int arow = m0 + lmod;
    v8f acc = {};
#pragma unroll
    for (int kb = 0; kb < 4; ++kb) {
      // A fragment: two 16B-aligned bf16 loads, no conversion needed
      const __bf16* __restrict__ wr = wbf + arow * CIN + kb * 32 + koff;
      const v8bf alo = *(const v8bf*)wr;         // K = kb*32 + koff + 0..7
      const v8bf ahi = *(const v8bf*)(wr + 16);  // K = kb*32 + koff + 16..23
      const v16bf a = __builtin_shufflevector(alo, ahi, 0, 1, 2, 3, 4, 5, 6, 7,
                                              8, 9, 10, 11, 12, 13, 14, 15);
      acc = __builtin_amdgcn_wmma_f32_16x16x32_bf16(
          /*neg_a=*/false, a, /*neg_b=*/false, bfrag[kb],
          /*c_mod=*/(short)0, acc, /*reuse_a=*/false, /*reuse_b=*/false);
    }
    // C layout: lane -> N = lane&15 (pixel), VGPR r -> M = r + 8*(lane>>4)
    float* __restrict__ op = qkv + (size_t)pcol * COUT + (m0 + (lhalf << 3));
#pragma unroll
    for (int r = 0; r < 8; ++r) op[r] = acc[r];  // 32B contiguous per lane
  }
}

// ---------------------------------------------------------------------------
// Stage 2: neighborhood attention. One thread per (pixel, head).
//   Zero-padded OOB neighbors contribute score == 0 to the softmax
//   (matches reference's zero-padded im2col) and V == 0 to the output.
// ---------------------------------------------------------------------------
__global__ __launch_bounds__(256) void natt_kernel(
    const float* __restrict__ qkv, float* __restrict__ out) {
  const int tid  = blockIdx.x * 256 + threadIdx.x;
  const int head = tid & (NHEADS - 1);
  const int p    = tid >> 3;
  const int n    = p >> 12;
  const int rem  = p & 4095;
  const int y    = rem >> 6;
  const int xx   = rem & 63;

  const float4* __restrict__ qv =
      (const float4*)(qkv + (size_t)p * COUT + head * HDIM);
  float4 q[8];
#pragma unroll
  for (int i = 0; i < 8; ++i) q[i] = qv[i];

  float sc[25];
#pragma unroll
  for (int j = 0; j < 25; ++j) {
    const int dy = j / 5 - 2, dx = j % 5 - 2;
    const int yy = y + dy, xn = xx + dx;
    float s = 0.0f;
    if ((unsigned)yy < (unsigned)HH && (unsigned)xn < (unsigned)WW) {
      const int kp = (n << 12) + (yy << 6) + xn;
      const float4* __restrict__ kv =
          (const float4*)(qkv + (size_t)kp * COUT + DMOD + head * HDIM);
      float d = 0.0f;
#pragma unroll
      for (int i = 0; i < 8; ++i) {
        const float4 k4 = kv[i];
        d = fmaf(q[i].x, k4.x, d); d = fmaf(q[i].y, k4.y, d);
        d = fmaf(q[i].z, k4.z, d); d = fmaf(q[i].w, k4.w, d);
      }
      s = d;
    }
    sc[j] = s;
  }

  float mx = sc[0];
#pragma unroll
  for (int j = 1; j < 25; ++j) mx = fmaxf(mx, sc[j]);
  float sum = 0.0f;
#pragma unroll
  for (int j = 0; j < 25; ++j) { sc[j] = __expf(sc[j] - mx); sum += sc[j]; }
  const float inv = 1.0f / sum;

  float4 acc[8] = {};
#pragma unroll
  for (int j = 0; j < 25; ++j) {
    const int dy = j / 5 - 2, dx = j % 5 - 2;
    const int yy = y + dy, xn = xx + dx;
    if ((unsigned)yy < (unsigned)HH && (unsigned)xn < (unsigned)WW) {
      const float wgt = sc[j] * inv;
      const int vp = (n << 12) + (yy << 6) + xn;
      const float4* __restrict__ vv =
          (const float4*)(qkv + (size_t)vp * COUT + 2 * DMOD + head * HDIM);
#pragma unroll
      for (int i = 0; i < 8; ++i) {
        const float4 v4 = vv[i];
        acc[i].x = fmaf(wgt, v4.x, acc[i].x);
        acc[i].y = fmaf(wgt, v4.y, acc[i].y);
        acc[i].z = fmaf(wgt, v4.z, acc[i].z);
        acc[i].w = fmaf(wgt, v4.w, acc[i].w);
      }
    }
  }

  // out layout (n, c, h, w): addr = (n*256 + c)*4096 + rem, c = head*32 + d
  float* __restrict__ ob = out + ((size_t)(n * DMOD + head * HDIM) << 12) + rem;
#pragma unroll
  for (int i = 0; i < 8; ++i) {
    ob[(size_t)(4 * i + 0) << 12] = acc[i].x;
    ob[(size_t)(4 * i + 1) << 12] = acc[i].y;
    ob[(size_t)(4 * i + 2) << 12] = acc[i].z;
    ob[(size_t)(4 * i + 3) << 12] = acc[i].w;
  }
}

extern "C" void kernel_launch(void* const* d_in, const int* in_sizes, int n_in,
                              void* d_out, int out_size, void* d_ws, size_t ws_size,
                              hipStream_t stream) {
  const float* x = (const float*)d_in[0];   // (4,128,64,64)
  const float* w = (const float*)d_in[1];   // (768,128)
  float*  qkv = (float*)d_ws;               // [16384][768] fp32 = 48 MB
  __bf16* wbf = (__bf16*)(qkv + (size_t)NPIX * COUT);  // 768*128 bf16 = 192 KB

  // Stage 0: convert weights to bf16 once (98304 elems, 4 per thread)
  wcvt_kernel<<<(COUT * CIN) / (256 * 4), 256, 0, stream>>>(w, wbf);
  // Stage 1: one block per 16-pixel tile; all 768 out-channels in-block
  qkv_gemm_wmma<<<NPIX / 16, 256, 0, stream>>>(x, wbf, qkv);
  // Stage 2: 16384 pixels * 8 heads = 131072 threads
  natt_kernel<<<512, 256, 0, stream>>>(qkv, (float*)d_out);
}
